// GPT_23021024706847
// MI455X (gfx1250) — compile-verified, matrix operand order checked by reading, not run on hardware
//
#include <hip/hip_runtime.h>
#include <hip/hip_bf16.h>
#include <math.h>

// ---------------------------------------------------------------------------
// GPT forward pass for gfx1250 (MI455X).
// GEMMs: v_wmma_f32_16x16x32_bf16, 64x64 tile per wave (16 WMMA / K-step),
// 128x256 block tile, LDS double-buffered with register-staged prefetch.
// fp32 -> bf16 conversion happens while staging into LDS (weights are read
// once per use, so on-the-fly conversion costs no extra HBM traffic).
// Attention K/V tiles are staged with gfx1250 async global->LDS copies
// (GLOBAL_LOAD_ASYNC_TO_LDS_B128, ASYNCcnt-tracked).
// LayerNorm / softmax / epilogues stay fp32.
// Model dims: B=2 T=1024 C=1024 H=16 D=64 L=8 HID=4096 V=32000, M=B*T=2048.
// All GEMM N are multiples of 256, M multiple of 128, K multiple of 32.
// ---------------------------------------------------------------------------

typedef __attribute__((ext_vector_type(16))) __bf16 bf16x16;
typedef __attribute__((ext_vector_type(8)))  float  f32x8;

#define TM 128     // block tile M
#define TBN 256    // block tile N
#define TK 32      // K step
#define LDPAD 40   // bf16 elements per LDS row (32 + 8 pad to stagger banks)

__device__ __forceinline__ void wait_asynccnt0() {
#if __has_builtin(__builtin_amdgcn_s_wait_asynccnt)
  __builtin_amdgcn_s_wait_asynccnt(0);
#else
  asm volatile("s_wait_asynccnt 0x0" ::: "memory");
#endif
}

// Async copy of 16 bytes global -> LDS (per lane). ldsaddr is the low 32 bits
// of the generic pointer; per the flat-aperture rules addr[31:0] is the LDS
// offset. Tracked with ASYNCcnt.
__device__ __forceinline__ void async_copy_b128(unsigned ldsaddr, const void* gaddr) {
  asm volatile("global_load_async_to_lds_b128 %0, %1, off"
               :: "v"(ldsaddr), "v"(gaddr) : "memory");
}

// Load one 16x32 bf16 WMMA operand fragment from LDS.
// Layout per CDNA5 ISA 7.12.2 (16-bit A-matrix 16x32):
//   lane%16 selects the row; lanes 0-15 cover K={0..7,16..23},
//   lanes 16-31 cover K={8..15,24..31}.
__device__ __forceinline__ bf16x16 load_frag(const __bf16* base, int lane) {
  const int r = lane & 15;
  const int h = (lane >> 4) << 3;       // 0 or 8
  const __bf16* p = base + r * LDPAD;
  bf16x16 f;
#pragma unroll
  for (int i = 0; i < 8; ++i) f[i] = p[h + i];
#pragma unroll
  for (int i = 0; i < 8; ++i) f[8 + i] = p[16 + h + i];
  return f;
}

// out[M,N] = act(A[M,K] @ B + bias) + resid
//   transB==0: Bw is [K,N] row-major.  transB==1: Bw is [N,K] row-major.
//   gelu!=0  : exact GELU (erf form) applied after bias, before residual.
__global__ __launch_bounds__(256)
void k_gemm(const float* __restrict__ A, const float* __restrict__ Bw,
            const float* __restrict__ bias, const float* __restrict__ resid,
            float* __restrict__ out, int M, int N, int K, int transB, int gelu)
{
  __shared__ __bf16 sA[2][TM * LDPAD];
  __shared__ __bf16 sB[2][TBN * LDPAD];

  const int tid  = threadIdx.x;
  const int lane = tid & 31;
  const int wave = tid >> 5;            // 8 waves: 2 along M x 4 along N
  const int mBase = blockIdx.y * TM;
  const int nBase = blockIdx.x * TBN;
  const int wm = (wave & 1) * 64;       // wave tile: 64 rows x 64 cols
  const int wn = (wave >> 1) * 64;

  // staging thread mappings
  const int rA  = tid >> 1, cA = (tid & 1) * 16;   // A: 16 floats/thread
  const int nBt = tid;                             // B transposed: 1 row (32 f)
  const int kB  = tid >> 3, nB0 = (tid & 7) * 32;  // B normal: 32 floats/thread

  float ra[16], rb[32];
  f32x8 acc[4][4] = {};

  auto g_load = [&](int k0) {
    const float* srcA = A + (size_t)(mBase + rA) * K + k0 + cA;
#pragma unroll
    for (int i = 0; i < 16; ++i) ra[i] = srcA[i];
    if (transB) {
      const float* srcB = Bw + (size_t)(nBase + nBt) * K + k0;
#pragma unroll
      for (int i = 0; i < 32; ++i) rb[i] = srcB[i];
    } else {
      const float* srcB = Bw + (size_t)(k0 + kB) * N + nBase + nB0;
#pragma unroll
      for (int i = 0; i < 32; ++i) rb[i] = srcB[i];
    }
  };
  auto s_store = [&](int buf) {
    __bf16* dA = &sA[buf][rA * LDPAD + cA];
#pragma unroll
    for (int i = 0; i < 16; ++i) dA[i] = (__bf16)ra[i];
    if (transB) {
      __bf16* dB = &sB[buf][nBt * LDPAD];
#pragma unroll
      for (int i = 0; i < 32; ++i) dB[i] = (__bf16)rb[i];
    } else {
#pragma unroll
      for (int i = 0; i < 32; ++i) sB[buf][(nB0 + i) * LDPAD + kB] = (__bf16)rb[i];
    }
  };

  g_load(0);
  s_store(0);
  __syncthreads();

  const int nk = K / TK;
  for (int kt = 0; kt < nk; ++kt) {
    const int buf = kt & 1;

    // prefetch tile kt+1 into registers (overlaps the WMMAs below)
    if (kt + 1 < nk) {
      g_load((kt + 1) * TK);
      // hint the tile after next toward the caches (global_prefetch_b8)
      if (kt + 2 < nk) {
        __builtin_prefetch(A + (size_t)(mBase + rA) * K + (kt + 2) * TK + cA, 0, 1);
        __builtin_prefetch(transB ? Bw + (size_t)(nBase + nBt) * K + (kt + 2) * TK
                                  : Bw + (size_t)((kt + 2) * TK + kB) * N + nBase + nB0,
                           0, 1);
      }
    }

    bf16x16 af[4], bfr[4];
#pragma unroll
    for (int mi = 0; mi < 4; ++mi)
      af[mi] = load_frag(&sA[buf][(wm + mi * 16) * LDPAD], lane);
#pragma unroll
    for (int ni = 0; ni < 4; ++ni)
      bfr[ni] = load_frag(&sB[buf][(wn + ni * 16) * LDPAD], lane);

#pragma unroll
    for (int mi = 0; mi < 4; ++mi)
#pragma unroll
      for (int ni = 0; ni < 4; ++ni)
        acc[mi][ni] = __builtin_amdgcn_wmma_f32_16x16x32_bf16(
            false, af[mi], false, bfr[ni], (short)0, acc[mi][ni], false, false);

    if (kt + 1 < nk) s_store(buf ^ 1);
    __syncthreads();
  }

  // --- epilogue: D element (m,n) lives in lane n+16*(m>=8), vgpr m%8 ---
  const int nl = lane & 15;
  const int mh = (lane >> 4) * 8;
#pragma unroll
  for (int mi = 0; mi < 4; ++mi) {
#pragma unroll
    for (int ni = 0; ni < 4; ++ni) {
#pragma unroll
      for (int r = 0; r < 8; ++r) {
        const int gm = mBase + wm + mi * 16 + mh + r;
        const int gn = nBase + wn + ni * 16 + nl;
        float v = acc[mi][ni][r];
        if (bias)  v += bias[gn];
        if (gelu)  v = 0.5f * v * (1.0f + erff(v * 0.70710678118654752f));
        if (resid) v += resid[(size_t)gm * N + gn];
        out[(size_t)gm * N + gn] = v;
      }
    }
  }
}

// x[row] = tok_emb[idx[row]] + pos_emb[row % T]
__global__ __launch_bounds__(256)
void k_embed(const int* __restrict__ idx, const float* __restrict__ tok,
             const float* __restrict__ pos, float* __restrict__ x,
             int T_, int Cc)
{
  const int row = blockIdx.x;
  const int t = row % T_;
  const int v = idx[row];
  const float* tr = tok + (size_t)v * Cc;
  const float* pr = pos + (size_t)t * Cc;
  float* xr = x + (size_t)row * Cc;
  for (int c = threadIdx.x; c < Cc; c += 256) xr[c] = tr[c] + pr[c];
}

// Row LayerNorm, one block per row.
__global__ __launch_bounds__(256)
void k_lnorm(const float* __restrict__ x, const float* __restrict__ g,
             const float* __restrict__ b, float* __restrict__ out, int Cc)
{
  __shared__ float red[256];
  const int row = blockIdx.x;
  const float* xr = x + (size_t)row * Cc;

  float s = 0.f;
  for (int c = threadIdx.x; c < Cc; c += 256) s += xr[c];
  red[threadIdx.x] = s;
  __syncthreads();
  for (int st = 128; st > 0; st >>= 1) {
    if (threadIdx.x < st) red[threadIdx.x] += red[threadIdx.x + st];
    __syncthreads();
  }
  const float mu = red[0] / Cc;
  __syncthreads();

  float v = 0.f;
  for (int c = threadIdx.x; c < Cc; c += 256) {
    const float d = xr[c] - mu;
    v += d * d;
  }
  red[threadIdx.x] = v;
  __syncthreads();
  for (int st = 128; st > 0; st >>= 1) {
    if (threadIdx.x < st) red[threadIdx.x] += red[threadIdx.x + st];
    __syncthreads();
  }
  const float rstd = rsqrtf(red[0] / Cc + 1e-5f);

  for (int c = threadIdx.x; c < Cc; c += 256)
    out[(size_t)row * Cc + c] = (xr[c] - mu) * rstd * g[c] + b[c];
}

// Causal attention, online softmax. One q-row per thread; K/V tiles staged in
// LDS via async global->LDS copies (ASYNCcnt).
// qkv layout: [B*T, 3C] with q|k|v each [H,D] per token.  att: [B*T, C].
#define AT_ROWS 128
#define AT_KT 64
__global__ __launch_bounds__(AT_ROWS)
void k_attn(const float* __restrict__ qkv, float* __restrict__ att,
            int T_, int Hn)
{
  __shared__ __align__(16) float sK[AT_KT * 64];
  __shared__ __align__(16) float sV[AT_KT * 64];

  const int bh = blockIdx.y;
  const int b = bh / Hn;
  const int h = bh % Hn;
  const int qi = blockIdx.x * AT_ROWS + threadIdx.x;
  const int Cc = Hn * 64;
  const int C3 = 3 * Cc;
  const float scale = 0.125f;           // D^-0.5, D=64

  float q[64], acc[64];
  const float* qrow = qkv + (size_t)(b * T_ + qi) * C3 + h * 64;
#pragma unroll
  for (int d = 0; d < 64; ++d) { q[d] = qrow[d] * scale; acc[d] = 0.f; }
  float m = -3.0e38f, lsum = 0.f;

  // per-thread staging slice: half a K row + half a V row (32 floats each)
  const int j    = threadIdx.x >> 1;
  const int half = (threadIdx.x & 1) * 32;
  const unsigned kdst = (unsigned)(uintptr_t)&sK[j * 64 + half];
  const unsigned vdst = (unsigned)(uintptr_t)&sV[j * 64 + half];

  const int kend = (blockIdx.x + 1) * AT_ROWS;   // causal: keys < kend
  for (int kb = 0; kb < kend; kb += AT_KT) {
    const float* kr = qkv + (size_t)(b * T_ + kb + j) * C3 + Cc + h * 64 + half;
    const float* vr = kr + Cc;
    // async global -> LDS tile copy, 16B granules
#pragma unroll
    for (int c = 0; c < 8; ++c) {
      async_copy_b128(kdst + c * 16, kr + c * 4);
      async_copy_b128(vdst + c * 16, vr + c * 4);
    }
    if (kb + AT_KT < kend) {             // hint next tile (global_prefetch_b8)
      __builtin_prefetch(kr + (size_t)AT_KT * C3, 0, 1);
      __builtin_prefetch(vr + (size_t)AT_KT * C3, 0, 1);
    }
    wait_asynccnt0();
    __syncthreads();

    const int jmax = min(AT_KT, qi - kb + 1);    // <=0 -> no valid keys
    for (int jj = 0; jj < jmax; ++jj) {
      float s = 0.f;
#pragma unroll
      for (int d = 0; d < 64; ++d) s += q[d] * sK[jj * 64 + d];
      const float mn   = fmaxf(m, s);
      const float corr = __expf(m - mn);
      const float p    = __expf(s - mn);
      lsum = lsum * corr + p;
#pragma unroll
      for (int d = 0; d < 64; ++d) acc[d] = acc[d] * corr + p * sV[jj * 64 + d];
      m = mn;
    }
    __syncthreads();
  }

  const float inv = 1.0f / lsum;
  float* orow = att + (size_t)(b * T_ + qi) * Cc + h * 64;
#pragma unroll
  for (int d = 0; d < 64; ++d) orow[d] = acc[d] * inv;
}

// ---------------------------------------------------------------------------
extern "C" void kernel_launch(void* const* d_in, const int* in_sizes, int n_in,
                              void* d_out, int out_size, void* d_ws, size_t ws_size,
                              hipStream_t stream) {
  (void)in_sizes; (void)n_in; (void)out_size; (void)ws_size;

  constexpr int Bn = 2, T = 1024, C = 1024, Hn = 16, L = 8;
  constexpr int HID = 4 * C, V = 32000, M = Bn * T, C3 = 3 * C;

  const int*   idx     = (const int*)  d_in[0];
  const float* tok_emb = (const float*)d_in[1];
  const float* pos_emb = (const float*)d_in[2];
  const float* qkv_w   = (const float*)d_in[3];
  const float* proj_w  = (const float*)d_in[4];
  const float* proj_b  = (const float*)d_in[5];
  const float* ln1_g   = (const float*)d_in[6];
  const float* ln1_b   = (const float*)d_in[7];
  const float* ln2_g   = (const float*)d_in[8];
  const float* ln2_b   = (const float*)d_in[9];
  const float* fc1_w   = (const float*)d_in[10];
  const float* fc1_b   = (const float*)d_in[11];
  const float* fc2_w   = (const float*)d_in[12];
  const float* fc2_b   = (const float*)d_in[13];
  const float* lnf_g   = (const float*)d_in[14];
  const float* lnf_b   = (const float*)d_in[15];
  float* logits = (float*)d_out;

  // workspace layout (fp32): x | h | qkv | att | ff1  (~84 MB total)
  float* x    = (float*)d_ws;
  float* h    = x   + (size_t)M * C;
  float* qkv  = h   + (size_t)M * C;
  float* att  = qkv + (size_t)M * C3;
  float* ff1  = att + (size_t)M * C;

  dim3 blk(256);
  k_embed<<<M, blk, 0, stream>>>(idx, tok_emb, pos_emb, x, T, C);

  for (int l = 0; l < L; ++l) {
    const float* qw  = qkv_w  + (size_t)l * C * C3;
    const float* pw  = proj_w + (size_t)l * C * C;
    const float* pb  = proj_b + (size_t)l * C;
    const float* w1  = fc1_w  + (size_t)l * C * HID;
    const float* bb1 = fc1_b  + (size_t)l * HID;
    const float* w2  = fc2_w  + (size_t)l * HID * C;
    const float* bb2 = fc2_b  + (size_t)l * C;

    // h = LN1(x);  qkv = h @ qkv_w
    k_lnorm<<<M, blk, 0, stream>>>(x, ln1_g + (size_t)l * C, ln1_b + (size_t)l * C, h, C);
    k_gemm<<<dim3(C3 / TBN, M / TM), blk, 0, stream>>>(h, qw, nullptr, nullptr, qkv,
                                                       M, C3, C, 0, 0);
    // att = causal_attention(qkv)
    k_attn<<<dim3(T / AT_ROWS, Bn * Hn), dim3(AT_ROWS), 0, stream>>>(qkv, att, T, Hn);
    // x = x + att @ proj_w + proj_b
    k_gemm<<<dim3(C / TBN, M / TM), blk, 0, stream>>>(att, pw, pb, x, x, M, C, C, 0, 0);
    // h = LN2(x);  ff1 = gelu(h @ w1 + b1);  x = x + ff1 @ w2 + b2
    k_lnorm<<<M, blk, 0, stream>>>(x, ln2_g + (size_t)l * C, ln2_b + (size_t)l * C, h, C);
    k_gemm<<<dim3(HID / TBN, M / TM), blk, 0, stream>>>(h, w1, bb1, nullptr, ff1,
                                                        M, HID, C, 0, 1);
    k_gemm<<<dim3(C / TBN, M / TM), blk, 0, stream>>>(ff1, w2, bb2, x, x, M, C, HID, 0, 0);
  }

  // final LN + tied lm_head (B stored [V, C] -> transB)
  k_lnorm<<<M, blk, 0, stream>>>(x, lnf_g, lnf_b, h, C);
  k_gemm<<<dim3(V / TBN, M / TM), blk, 0, stream>>>(h, tok_emb, nullptr, nullptr, logits,
                                                    M, V, C, 1, 0);
}
